// Attention_27968827031909
// MI455X (gfx1250) — compile-verified, matrix operand order checked by reading, not run on hardware
//
#include <hip/hip_runtime.h>
#include <math.h>
#include <stdint.h>

// ---------------------------------------------------------------------------
// Sizes (fixed by the reference)
// ---------------------------------------------------------------------------
#define BB   8
#define FF   512
#define SS   8192
#define HH   8
#define DD   64
#define MM   (BB * SS)          // 65536 GEMM rows
#define NG   (BB * 32)          // 256 groupnorm (b,g) pairs
#define GELEMS (16 * SS)        // 131072 elements per (b,g) region

typedef __bf16 bf16;
typedef __attribute__((ext_vector_type(16))) bf16  v16bf;
typedef __attribute__((ext_vector_type(8)))  bf16  v8bf;
typedef __attribute__((ext_vector_type(8)))  float v8f;
typedef __attribute__((ext_vector_type(2)))  float v2f;
typedef __attribute__((ext_vector_type(4)))  uint32_t u32x4;
typedef __attribute__((ext_vector_type(4)))  int      i32x4;
typedef __attribute__((ext_vector_type(8)))  int      i32x8;

#if defined(__has_builtin)
#if __has_builtin(__builtin_amdgcn_wmma_f32_16x16x4_f32)
#define HAVE_WMMA_F32X4 1
#endif
#endif

__device__ __forceinline__ v8f zero8() {
  v8f z;
#pragma unroll
  for (int i = 0; i < 8; i++) z[i] = 0.f;
  return z;
}

__device__ __forceinline__ v16bf comb16(v8bf lo, v8bf hi) {
  v16bf r;
#pragma unroll
  for (int i = 0; i < 8; i++) { r[i] = lo[i]; r[i + 8] = hi[i]; }
  return r;
}

// ---------------------------------------------------------------------------
// TDM: move a 128-row x 64B tile (bf16, row stride 512 elems) into LDS.
// D# pad feature inserts 20 DWORDs after every 16 DWORDs stored -> effective
// LDS row stride 144B, matching the compute-side LDSS=72 bf16 layout.
// Group0: count=1 | lds_addr | global_addr[56:0] | type=2
// Group1: data_size=2B, pad_enable, pad_interval=16DW(code 3), pad_amount=20DW
//         (code 19), tensor_dim0=512, tensor_dim1=1M, tile=32x128, stride=512
// ---------------------------------------------------------------------------
__device__ __forceinline__ void tdm_load_tile_bf16(uint32_t lds_off,
                                                   const bf16* gptr) {
  uint64_t ga = (uint64_t)(uintptr_t)gptr;
  u32x4 g0;
  g0[0] = 1u;                                                  // count=1
  g0[1] = lds_off;                                             // lds_addr
  g0[2] = (uint32_t)ga;                                        // addr[31:0]
  g0[3] = ((uint32_t)(ga >> 32) & 0x01FFFFFFu) | 0x80000000u;  // addr[56:32]|type=2
  i32x8 g1;
  g1[0] = (int)((1u << 16) | (1u << 20) | (3u << 22) | (19u << 25));
  g1[1] = (int)(512u << 16);            // tensor_dim0 = 512 (bits 63:48)
  g1[2] = 0;                            // tensor_dim0 hi / tensor_dim1 lo
  g1[3] = (int)(0x10u | (32u << 16));   // tensor_dim1 hi (=1M rows), tile_dim0=32
  g1[4] = 128;                          // tile_dim1 = 128, tile_dim2 = 0
  g1[5] = 512;                          // tensor_dim0_stride = 512 elems
  g1[6] = 0;
  g1[7] = 0;
  i32x4 z4 = {0, 0, 0, 0};
#if defined(__clang_major__) && (__clang_major__ >= 23)
  i32x8 z8 = {0, 0, 0, 0, 0, 0, 0, 0};
  __builtin_amdgcn_tensor_load_to_lds(g0, g1, z4, z4, z8, 0);
#else
  __builtin_amdgcn_tensor_load_to_lds(g0, g1, z4, z4, 0);
#endif
}

// ---------------------------------------------------------------------------
// Weight downconvert: qkv_w [1536,512] f32 -> bf16, out_proj [512,512] -> bf16
// ---------------------------------------------------------------------------
__global__ __launch_bounds__(256) void cvt_weights_kernel(
    const float* __restrict__ w, const float* __restrict__ p,
    bf16* __restrict__ wb, bf16* __restrict__ pb) {
  int i = blockIdx.x * 256 + threadIdx.x;
  if (i < 1536 * 512) wb[i] = (bf16)w[i];
  if (i < 512 * 512)  pb[i] = (bf16)p[i];
}

// ---------------------------------------------------------------------------
// GroupNorm stats: one block per (b,g); region is 131072 contiguous floats.
// ---------------------------------------------------------------------------
__global__ __launch_bounds__(256) void gn_stats_kernel(
    const float* __restrict__ x, float* __restrict__ mu, float* __restrict__ rs) {
  const float4* p = (const float4*)(x + (size_t)blockIdx.x * GELEMS);
  float s = 0.f, s2 = 0.f;
  for (int i = threadIdx.x; i < GELEMS / 4; i += 256) {
    float4 v = p[i];
    s  += v.x + v.y + v.z + v.w;
    s2 += v.x * v.x + v.y * v.y + v.z * v.z + v.w * v.w;
  }
  __shared__ float sa[256], sb[256];
  sa[threadIdx.x] = s; sb[threadIdx.x] = s2;
  __syncthreads();
  for (int off = 128; off > 0; off >>= 1) {
    if ((int)threadIdx.x < off) {
      sa[threadIdx.x] += sa[threadIdx.x + off];
      sb[threadIdx.x] += sb[threadIdx.x + off];
    }
    __syncthreads();
  }
  if (threadIdx.x == 0) {
    float m   = sa[0] * (1.f / GELEMS);
    float var = sb[0] * (1.f / GELEMS) - m * m;
    mu[blockIdx.x] = m;
    rs[blockIdx.x] = rsqrtf(var + 1e-8f);
  }
}

// ---------------------------------------------------------------------------
// GN1 normalize + transpose: x[B,F,S] f32 -> xn[(b*S+s), F] bf16
// ---------------------------------------------------------------------------
__global__ __launch_bounds__(256) void gn_norm_t_kernel(
    const float* __restrict__ x, const float* __restrict__ mu,
    const float* __restrict__ rs, const float* __restrict__ gamma,
    const float* __restrict__ beta, bf16* __restrict__ xn) {
  __shared__ float tile[32][33];
  int b = blockIdx.z, f0 = blockIdx.y * 32, s0 = blockIdx.x * 32;
  int tx = threadIdx.x, ty = threadIdx.y;
#pragma unroll
  for (int i = 0; i < 4; i++) {
    int fl = ty + i * 8;
    tile[fl][tx] = x[((size_t)b * FF + f0 + fl) * SS + s0 + tx];
  }
  __syncthreads();
  int f = f0 + tx;
  int bg = b * 32 + (f >> 4);
  float m = mu[bg], r = rs[bg];
  float a = r * gamma[f];
  float c = beta[f] - m * a;
#pragma unroll
  for (int i = 0; i < 4; i++) {
    int sl = ty + i * 8;
    float v = tile[tx][sl] * a + c;
    xn[((size_t)b * SS + s0 + sl) * FF + f] = (bf16)v;
  }
}

// ---------------------------------------------------------------------------
// bf16 WMMA GEMM: C[M,N] = A[M,K] * B[N,K]^T   (B row-major over K)
// Block tile 128x128, 8 waves of 64x32, K step 32.
// A/B tiles are DMA'd into LDS by the Tensor Data Mover (waves 0 and 1),
// synchronized with s_wait_tensorcnt + workgroup barrier; the remaining waves
// never touch the memory pipe in the mainloop.
// ---------------------------------------------------------------------------
#define LDSS 72   // LDS row stride in bf16 (64B data + 80B TDM pad = 144B)

template <int MODE>
__global__ __launch_bounds__(256) void gemm_bf16_kernel(
    const bf16* __restrict__ A, const bf16* __restrict__ Bm, int K,
    const float* __restrict__ bias,
    float* __restrict__ qo, float* __restrict__ ko, float* __restrict__ vo,
    float* __restrict__ yo) {
  __shared__ __align__(16) bf16 lA[128 * LDSS];
  __shared__ __align__(16) bf16 lB[128 * LDSS];
  const int tid  = threadIdx.x;
  const int wave = tid >> 5, lane = tid & 31;
  const int wm = wave >> 2, wn = wave & 3;         // 2 x 4 wave grid
  const int lrow = lane & 15, lhalf = lane >> 4;
  const int m_blk = blockIdx.x * 128;
  const int n_blk = blockIdx.y * 128;
  const uint32_t lA_off = (uint32_t)(uintptr_t)(void*)lA;
  const uint32_t lB_off = (uint32_t)(uintptr_t)(void*)lB;

  v8f acc[4][2];
#pragma unroll
  for (int i = 0; i < 4; i++)
#pragma unroll
    for (int j = 0; j < 2; j++) acc[i][j] = zero8();

  for (int k0 = 0; k0 < K; k0 += 32) {
    if (wave == 0)
      tdm_load_tile_bf16(lA_off, A + (size_t)m_blk * K + k0);
    else if (wave == 1)
      tdm_load_tile_bf16(lB_off, Bm + (size_t)n_blk * K + k0);
    if (wave < 2) __builtin_amdgcn_s_wait_tensorcnt(0);
    __syncthreads();

    v16bf af[4], bfg[2];
#pragma unroll
    for (int mi = 0; mi < 4; mi++) {
      const bf16* p = lA + (wm * 64 + mi * 16 + lrow) * LDSS + lhalf * 8;
      af[mi] = comb16(*(const v8bf*)p, *(const v8bf*)(p + 16));
    }
#pragma unroll
    for (int ni = 0; ni < 2; ni++) {
      const bf16* p = lB + (wn * 32 + ni * 16 + lrow) * LDSS + lhalf * 8;
      bfg[ni] = comb16(*(const v8bf*)p, *(const v8bf*)(p + 16));
    }
#pragma unroll
    for (int mi = 0; mi < 4; mi++)
#pragma unroll
      for (int ni = 0; ni < 2; ni++)
        acc[mi][ni] = __builtin_amdgcn_wmma_f32_16x16x32_bf16(
            false, af[mi], false, bfg[ni], (short)0, acc[mi][ni], false, false);
    __syncthreads();
  }

  // epilogue: element acc[mi][ni][r] sits at (m_local = mi*16 + lhalf*8 + r,
  //                                           n_local = ni*16 + lrow)
#pragma unroll
  for (int mi = 0; mi < 4; mi++) {
#pragma unroll
    for (int ni = 0; ni < 2; ni++) {
#pragma unroll
      for (int r = 0; r < 8; r++) {
        int m = m_blk + wm * 64 + mi * 16 + lhalf * 8 + r;
        int n = n_blk + wn * 32 + ni * 16 + lrow;
        float val = acc[mi][ni][r];
        if (MODE == 0) {
          int i3 = n >> 9, c = n & 511;
          val += bias[(i3 << 9) + c];
          if (i3 < 2) val = (val > 0.f) ? (val + 1.f) : __expf(val);  // elu+1
          float* dst = (i3 == 0) ? qo : (i3 == 1) ? ko : vo;
          dst[(size_t)m * FF + c] = val;
        } else {
          int b = m >> 13, l = m & (SS - 1);
          yo[(((size_t)b << 9) + n) * SS + l] = val;
        }
      }
    }
  }
}

// ---------------------------------------------------------------------------
// k_sum[b, c] = sum_l k[(b*S+l), c]
// ---------------------------------------------------------------------------
__global__ __launch_bounds__(256) void ksum_kernel(
    const float* __restrict__ k, float* __restrict__ ks) {
  int b = blockIdx.x >> 1;
  int c = ((blockIdx.x & 1) << 8) + threadIdx.x;
  const float* p = k + (size_t)b * SS * FF + c;
  float s = 0.f;
  for (int l = 0; l < SS; l++) s += p[(size_t)l * FF];
  ks[b * FF + c] = s;
}

// ---------------------------------------------------------------------------
// KV[b,h,dv,dk] = sum_l v[l,dv] * k[l,dk]  via v_wmma_f32_16x16x4_f32.
// k/v tiles staged global->LDS with ASYNC loads (no VGPR round-trip),
// synchronized with s_wait_asynccnt + barrier.  LDS rows padded to 68
// dwords (272B) to keep 16B alignment for the B128 async writes.
// ---------------------------------------------------------------------------
__global__ __launch_bounds__(256) void kv_kernel(
    const float* __restrict__ kbuf, const float* __restrict__ vbuf,
    float* __restrict__ kv) {
  int b = blockIdx.x >> 3, h = blockIdx.x & 7;
  __shared__ __align__(16) float kt[64][68];
  __shared__ __align__(16) float vt[64][68];
  int tid = threadIdx.x, wave = tid >> 5, lane = tid & 31;
  int lrow = lane & 15, lhalf = lane >> 4;
  const uint32_t kt_off = (uint32_t)(uintptr_t)(void*)&kt[0][0];
  const uint32_t vt_off = (uint32_t)(uintptr_t)(void*)&vt[0][0];

#if HAVE_WMMA_F32X4
  v8f acc[2] = {zero8(), zero8()};
#else
  float facc[16];
#pragma unroll
  for (int i = 0; i < 16; i++) facc[i] = 0.f;
  int fdv = tid >> 2, fdk0 = (tid & 3) * 16;
#endif

  for (int l0 = 0; l0 < SS; l0 += 64) {
#pragma unroll
    for (int j = 0; j < 4; j++) {
      int cidx = tid + j * 256;              // 0..1023 -> 64 rows x 16 chunks
      int r = cidx >> 4, cc = (cidx & 15) << 2;
      size_t gelem = ((size_t)(b * SS + l0 + r)) * FF + h * DD + cc;
      uint64_t gk = (uint64_t)(uintptr_t)(kbuf + gelem);
      uint64_t gv = (uint64_t)(uintptr_t)(vbuf + gelem);
      uint32_t lk = kt_off + (uint32_t)(r * 68 + cc) * 4u;
      uint32_t lv = vt_off + (uint32_t)(r * 68 + cc) * 4u;
      asm volatile("global_load_async_to_lds_b128 %0, %1, off"
                   :: "v"(lk), "v"(gk) : "memory");
      asm volatile("global_load_async_to_lds_b128 %0, %1, off"
                   :: "v"(lv), "v"(gv) : "memory");
    }
    asm volatile("s_wait_asynccnt 0x0" ::: "memory");
    __syncthreads();

#if HAVE_WMMA_F32X4
    for (int kk = 0; kk < 64; kk += 4) {
      int l = kk + lhalf * 2;
#pragma unroll
      for (int j = 0; j < 2; j++) {
        int t = wave * 2 + j;
        int mt = t >> 2, nt = t & 3;
        v2f a, bb;
        a[0]  = vt[l][mt * 16 + lrow];       // A = V^T : A[dv, l]
        a[1]  = vt[l + 1][mt * 16 + lrow];
        bb[0] = kt[l][nt * 16 + lrow];       // B = K   : B[l, dk]
        bb[1] = kt[l + 1][nt * 16 + lrow];
        acc[j] = __builtin_amdgcn_wmma_f32_16x16x4_f32(
            false, a, false, bb, (short)0, acc[j], false, false);
      }
    }
#else
    for (int l = 0; l < 64; l++) {
      float vv = vt[l][fdv];
#pragma unroll
      for (int i = 0; i < 16; i++) facc[i] += vv * kt[l][fdk0 + i];
    }
#endif
    __syncthreads();
  }

  float* base = kv + ((size_t)b * HH + h) * DD * DD;
#if HAVE_WMMA_F32X4
#pragma unroll
  for (int j = 0; j < 2; j++) {
    int t = wave * 2 + j;
    int mt = t >> 2, nt = t & 3;
#pragma unroll
    for (int r = 0; r < 8; r++) {
      int dv = mt * 16 + lhalf * 8 + r;
      int dk = nt * 16 + lrow;
      base[dv * DD + dk] = acc[j][r];
    }
  }
#else
#pragma unroll
  for (int i = 0; i < 16; i++) base[fdv * DD + fdk0 + i] = facc[i];
#endif
}

// ---------------------------------------------------------------------------
// Attention output: per (b,l): z_h = q_h . ksum_h ; out = (q . KV^T) / z
// ---------------------------------------------------------------------------
__global__ __launch_bounds__(256) void attn_out_kernel(
    const float* __restrict__ q, const float* __restrict__ ks,
    const float* __restrict__ kv, bf16* __restrict__ ao) {
  int bl = blockIdx.x;
  int b  = bl >> 13;
  __shared__ float qs[FF];
  __shared__ float zs[HH];
  int tid = threadIdx.x;
  const float* qp = q + (size_t)bl * FF;
  qs[tid]       = qp[tid];
  qs[tid + 256] = qp[tid + 256];
  __syncthreads();

  int h = tid >> 5, lane = tid & 31;
  const float* ksh = ks + b * FF + h * DD;
  float part = qs[h * DD + lane] * ksh[lane] +
               qs[h * DD + 32 + lane] * ksh[32 + lane];
  for (int off = 16; off > 0; off >>= 1) part += __shfl_down(part, off);
  if (lane == 0) zs[h] = part;
  __syncthreads();

#pragma unroll
  for (int j = 0; j < 2; j++) {
    int o = tid + j * 256;
    int hh = o >> 6, dv = o & 63;
    const float* kvp = kv + (((size_t)b * HH + hh) * DD + dv) * DD;
    const float* qh  = qs + hh * DD;
    float s = 0.f;
#pragma unroll 8
    for (int f = 0; f < DD; f++) s += qh[f] * kvp[f];
    ao[(size_t)bl * FF + o] = (bf16)(s / zs[hh]);
  }
}

// ---------------------------------------------------------------------------
// Final: out = x + gate * (GN2(y));   float4-vectorized over [B,F,S]
// ---------------------------------------------------------------------------
__global__ __launch_bounds__(256) void final_kernel(
    const float* __restrict__ x, const float* __restrict__ y,
    const float* __restrict__ mu, const float* __restrict__ rs,
    const float* __restrict__ g2, const float* __restrict__ b2,
    const float* __restrict__ gate, float* __restrict__ out) {
  size_t i4  = (size_t)blockIdx.x * 256 + threadIdx.x;
  size_t idx = i4 * 4;
  int f = (int)((idx >> 13) & 511);
  int b = (int)(idx >> 22);
  int bg = b * 32 + (f >> 4);
  float m = mu[bg], r = rs[bg];
  float a = r * g2[f];
  float c = b2[f] - m * a;
  float gt = gate[f];
  float4 yv = ((const float4*)y)[i4];
  float4 xv = ((const float4*)x)[i4];
  float4 o;
  o.x = xv.x + gt * (yv.x * a + c);
  o.y = xv.y + gt * (yv.y * a + c);
  o.z = xv.z + gt * (yv.z * a + c);
  o.w = xv.w + gt * (yv.w * a + c);
  ((float4*)out)[i4] = o;
}

// ---------------------------------------------------------------------------
// Host launcher
// ---------------------------------------------------------------------------
extern "C" void kernel_launch(void* const* d_in, const int* in_sizes, int n_in,
                              void* d_out, int out_size, void* d_ws, size_t ws_size,
                              hipStream_t stream) {
  const float* x        = (const float*)d_in[0];
  const float* qkv_w    = (const float*)d_in[1];
  const float* qkv_b    = (const float*)d_in[2];
  const float* out_proj = (const float*)d_in[3];
  const float* g1       = (const float*)d_in[4];
  const float* b1       = (const float*)d_in[5];
  const float* g2       = (const float*)d_in[6];
  const float* b2       = (const float*)d_in[7];
  const float* gate     = (const float*)d_in[8];
  float* out = (float*)d_out;

  char* wsb = (char*)d_ws;
  size_t off = 0;
  auto alloc = [&](size_t bytes) -> void* {
    void* p = wsb + off;
    off += (bytes + 255) & ~(size_t)255;
    return p;
  };
  bf16*  wb  = (bf16*)alloc((size_t)1536 * 512 * 2);       // qkv_w bf16
  bf16*  pb  = (bf16*)alloc((size_t)512 * 512 * 2);        // out_proj bf16
  bf16*  xn  = (bf16*)alloc((size_t)MM * FF * 2);          // normalized x, [M,F]
  float* q   = (float*)alloc((size_t)MM * FF * 4);
  float* k   = (float*)alloc((size_t)MM * FF * 4);
  float* v   = (float*)alloc((size_t)MM * FF * 4);
  float* ksm = (float*)alloc((size_t)BB * FF * 4);
  float* kv  = (float*)alloc((size_t)BB * HH * DD * DD * 4);
  bf16*  ao  = (bf16*)alloc((size_t)MM * FF * 2);          // attn out, [M,F]
  float* y   = (float*)alloc((size_t)BB * FF * SS * 4);    // pre-GN2
  float* mu1 = (float*)alloc(NG * 4);
  float* rs1 = (float*)alloc(NG * 4);
  float* mu2 = (float*)alloc(NG * 4);
  float* rs2 = (float*)alloc(NG * 4);

  cvt_weights_kernel<<<3072, 256, 0, stream>>>(qkv_w, out_proj, wb, pb);
  gn_stats_kernel<<<NG, 256, 0, stream>>>(x, mu1, rs1);
  gn_norm_t_kernel<<<dim3(SS / 32, FF / 32, BB), dim3(32, 8), 0, stream>>>(
      x, mu1, rs1, g1, b1, xn);
  gemm_bf16_kernel<0><<<dim3(MM / 128, 1536 / 128), 256, 0, stream>>>(
      xn, wb, FF, qkv_b, q, k, v, nullptr);
  ksum_kernel<<<BB * 2, 256, 0, stream>>>(k, ksm);
  kv_kernel<<<BB * HH, 256, 0, stream>>>(k, v, kv);
  attn_out_kernel<<<MM, 256, 0, stream>>>(q, ksm, kv, ao);
  gemm_bf16_kernel<1><<<dim3(MM / 128, FF / 128), 256, 0, stream>>>(
      ao, pb, FF, nullptr, nullptr, nullptr, nullptr, y);
  gn_stats_kernel<<<NG, 256, 0, stream>>>(y, mu2, rs2);
  final_kernel<<<(BB * FF * SS / 4) / 256, 256, 0, stream>>>(
      x, y, mu2, rs2, g2, b2, gate, out);
}
